// m01_55946243997756
// MI455X (gfx1250) — matrix operand order repaired and checked by reading, not compiled, not run on hardware
//
#include <hip/hip_runtime.h>
#include <hip/hip_bf16.h>

typedef __attribute__((ext_vector_type(16))) __bf16 v16bf;
typedef __attribute__((ext_vector_type(8)))  float  v8f;

#define Bn 8
#define Tn 20
#define Hn 96
#define Wn 96
#define Cn 16
#define Fn 16
#define HO 94
#define WO 94
#define NPIX (Bn * HO * WO)   /* 70688 */
#define KTOT 288              /* 144 (x patch) + 144 (h patch) = 9 * 32 */
#define KP   296              /* padded LDS K-stride: 592B rows, 16B aligned, conflict-free */
#define BM   64               /* pixels per workgroup */
#define NTHREADS 128          /* 4 waves */

union Frag { v16bf v; uint4 q[2]; };

static __device__ __forceinline__ unsigned short f2bf_bits(float f) {
  union { float f; unsigned u; } v; v.f = f;
  unsigned u = v.u;
  u += 0x7FFFu + ((u >> 16) & 1u);   // round-to-nearest-even
  return (unsigned short)(u >> 16);
}

__global__ __launch_bounds__(NTHREADS)
void convlstm_step(const float* __restrict__ x,            // (B,T,96,96,16) f32
                   const float* __restrict__ Wk,           // (3,3,16,64) f32
                   const float* __restrict__ Wr,           // (3,3,16,64) f32
                   const float* __restrict__ bias,         // (64) f32
                   const unsigned short* __restrict__ h_in,// (B,94,94,16) bf16
                   unsigned short* __restrict__ h_out,     // (B,94,94,16) bf16
                   float* __restrict__ c_state,            // (B,94,94,16) f32 (in-place)
                   float* __restrict__ out_f32,            // final h, f32
                   int t, int write_out)
{
  __shared__ unsigned short Alds[BM * KP];   // 37,888 B
  __shared__ unsigned short Blds[64 * KP];   // 37,888 B  (fused [Wk;Wr], col-major per n)

  const int tid  = threadIdx.x;
  const int lane = tid & 31;
  const int wave = tid >> 5;

  // ---- fused weight tile: Blds[n*KP + k] = bf16(W[k][n]), k<144 -> Wk, else Wr
  for (int idx = tid; idx < 64 * KTOT; idx += NTHREADS) {
    int n = idx & 63;
    int k = idx >> 6;
    float wv = (k < 144) ? Wk[k * 64 + n] : Wr[(k - 144) * 64 + n];
    Blds[n * KP + k] = f2bf_bits(wv);
  }

  // ---- A tile: 64 pixels x 288 (x_t VALID patch ++ h SAME patch)
  const int pix0 = blockIdx.x * BM;
  #pragma unroll
  for (int i = 0; i < (BM * 18) / NTHREADS; ++i) {   // 9 chunks of 16 elems per thread
    int q = tid + i * NTHREADS;                      // (pixel, source-tap) chunk id
    int m = q / 18;
    int s = q - m * 18;                              // 0..8: x taps, 9..17: h taps
    int p = pix0 + m;
    if (p >= NPIX) p = NPIX - 1;                     // clamp; garbage rows never stored
    int b  = p / (HO * WO);
    int rr = p - b * (HO * WO);
    int oy = rr / WO;
    int ox = rr - oy * WO;
    int tap = (s < 9) ? s : (s - 9);
    int ky = tap / 3, kx = tap - ky * 3;

    unsigned short tmp[16];
    if (s < 9) {
      // input conv patch: VALID over 96x96 -> always in bounds
      const float* src = x + ((((size_t)b * Tn + t) * Hn + (oy + ky)) * Wn + (ox + kx)) * Cn;
      #pragma unroll
      for (int c2 = 0; c2 < 16; ++c2) tmp[c2] = f2bf_bits(src[c2]);
    } else {
      // recurrent conv patch: SAME over 94x94, zero-pad halo
      int y = oy + ky - 1, xw = ox + kx - 1;
      if ((unsigned)y < (unsigned)HO && (unsigned)xw < (unsigned)WO) {
        const unsigned short* src = h_in + (((size_t)b * HO + y) * WO + xw) * Fn;
        #pragma unroll
        for (int c2 = 0; c2 < 16; ++c2) tmp[c2] = src[c2];
      } else {
        #pragma unroll
        for (int c2 = 0; c2 < 16; ++c2) tmp[c2] = 0;
      }
    }
    int kbase = ((s < 9) ? 0 : 144) + tap * 16;      // 32B-aligned within a 592B row
    unsigned short* dst = Alds + m * KP + kbase;
    #pragma unroll
    for (int c2 = 0; c2 < 16; ++c2) dst[c2] = tmp[c2];
  }

  __syncthreads();

  // ---- WMMA: wave w computes pixels [w*16, w*16+16) x all 64 gate channels
  const int mrow = lane & 15;
  const int hi   = lane >> 4;
  const unsigned short* arow = Alds + (wave * 16 + mrow) * KP;
  const unsigned short* b0p  = Blds + (0 * 16 + mrow) * KP;
  const unsigned short* b1p  = Blds + (1 * 16 + mrow) * KP;
  const unsigned short* b2p  = Blds + (2 * 16 + mrow) * KP;
  const unsigned short* b3p  = Blds + (3 * 16 + mrow) * KP;

  v8f acc0 = {}, acc1 = {}, acc2 = {}, acc3 = {};
  #pragma unroll
  for (int kc = 0; kc < 9; ++kc) {
    Frag a;
    int koff = kc * 32 + hi * 16;      // A layout: lanes>=16 hold K+16 half
    a.q[0] = *(const uint4*)(arow + koff);
    a.q[1] = *(const uint4*)(arow + koff + 8);
    int kb = kc * 32 + hi * 16;        // B layout: lanes>=16 hold K+16 half
    Frag b0, b1, b2, b3;
    b0.q[0] = *(const uint4*)(b0p + kb); b0.q[1] = *(const uint4*)(b0p + kb + 8);
    b1.q[0] = *(const uint4*)(b1p + kb); b1.q[1] = *(const uint4*)(b1p + kb + 8);
    b2.q[0] = *(const uint4*)(b2p + kb); b2.q[1] = *(const uint4*)(b2p + kb + 8);
    b3.q[0] = *(const uint4*)(b3p + kb); b3.q[1] = *(const uint4*)(b3p + kb + 8);
    acc0 = __builtin_amdgcn_wmma_f32_16x16x32_bf16(false, a.v, false, b0.v, (short)0, acc0, false, false);
    acc1 = __builtin_amdgcn_wmma_f32_16x16x32_bf16(false, a.v, false, b1.v, (short)0, acc1, false, false);
    acc2 = __builtin_amdgcn_wmma_f32_16x16x32_bf16(false, a.v, false, b2.v, (short)0, acc2, false, false);
    acc3 = __builtin_amdgcn_wmma_f32_16x16x32_bf16(false, a.v, false, b3.v, (short)0, acc3, false, false);
  }

  // ---- LSTM gate math. Gate split i/f/g/o = N-tiles 0..3; this lane holds
  // channel j = mrow of pixels m = r + 8*hi across the 4 accumulators.
  const float bi = bias[mrow];
  const float bf = bias[16 + mrow];
  const float bg = bias[32 + mrow];
  const float bo = bias[48 + mrow];
  #pragma unroll
  for (int r = 0; r < 8; ++r) {
    int m = wave * 16 + r + 8 * hi;
    int p = pix0 + m;
    if (p < NPIX) {
      float ig = fmaxf(acc0[r] + bi, 0.0f);   // recurrent_activation = relu
      float fg = fmaxf(acc1[r] + bf, 0.0f);
      float gg = tanhf(acc2[r] + bg);          // activation = tanh
      float og = fmaxf(acc3[r] + bo, 0.0f);
      size_t ci = (size_t)p * Fn + mrow;
      float cold = c_state[ci];
      float cnew = fg * cold + ig * gg;
      float hnew = og * tanhf(cnew);
      c_state[ci] = cnew;
      h_out[ci] = f2bf_bits(hnew);
      if (write_out) out_f32[ci] = hnew;
    }
  }
}

extern "C" void kernel_launch(void* const* d_in, const int* in_sizes, int n_in,
                              void* d_out, int out_size, void* d_ws, size_t ws_size,
                              hipStream_t stream) {
  (void)in_sizes; (void)n_in; (void)out_size; (void)ws_size;
  const float* x    = (const float*)d_in[0];
  const float* Wk   = (const float*)d_in[1];
  const float* Wr   = (const float*)d_in[2];
  const float* bias = (const float*)d_in[3];

  const size_t nState = (size_t)NPIX * Fn;                 // 1,131,008 elems
  float*          c_state = (float*)d_ws;                  // f32
  unsigned short* h0 = (unsigned short*)((char*)d_ws + nState * 4);
  unsigned short* h1 = (unsigned short*)((char*)d_ws + nState * 4 + nState * 2);

  // zero initial c and h0 (graph-capturable)
  hipMemsetAsync(d_ws, 0, nState * 4 + nState * 2, stream);

  dim3 grid((NPIX + BM - 1) / BM);   // 1105 workgroups
  dim3 block(NTHREADS);
  for (int t = 0; t < Tn; ++t) {
    const unsigned short* hin  = (t & 1) ? h1 : h0;
    unsigned short*       hout = (t & 1) ? h0 : h1;
    convlstm_step<<<grid, block, 0, stream>>>(x, Wk, Wr, bias, hin, hout,
                                              c_state, (float*)d_out,
                                              t, (t == Tn - 1) ? 1 : 0);
  }
}
